// MultiHeadAttentionWithContext_11656541241514
// MI455X (gfx1250) — matrix operand order repaired and checked by reading, not compiled, not run on hardware
//
#include <hip/hip_runtime.h>
#include <hip/hip_bf16.h>

// ---------------------------------------------------------------------------
// Fused MHA block for gfx1250 (MI455X): f16 WMMA everywhere, flash-attention.
//   D=1024, H=16, dk=64, B=4, S=2048, N=B*S=8192
// ---------------------------------------------------------------------------

typedef __attribute__((ext_vector_type(16))) _Float16 v16h;
typedef __attribute__((ext_vector_type(8)))  _Float16 v8h;
typedef __attribute__((ext_vector_type(8)))  float    v8f;

#define D_MODEL 1024
#define SEQ     2048
#define BATCH   4
#define NHEAD   16
#define DK      64
#define NTOK    (BATCH * SEQ)   // 8192

static __device__ __forceinline__ v8h ld8(const _Float16* p) {
    return *(const v8h*)p;
}

static __device__ __forceinline__ v16h combine16(v8h lo, v8h hi) {
    v16h r;
#pragma unroll
    for (int i = 0; i < 8; ++i) { r[i] = lo[i]; r[i + 8] = hi[i]; }
    return r;
}

static __device__ __forceinline__ v8f wmma_f16(v16h a, v16h b, v8f c) {
    // D = A(16x32 f16) * B(32x16 f16) + C(16x16 f32)
    return __builtin_amdgcn_wmma_f32_16x16x32_f16(
        false, a, false, b, (short)0, c, false, false);
}

// A-fragment: 16x32 tile of row-major X (ld elements/row), tile origin (row0,k0).
// lane m=lane&15 holds row row0+m; halves 0..7 = K k0+8*sel.., halves 8..15 = K k0+16+8*sel..
static __device__ __forceinline__ v16h load_a_frag(const _Float16* X, int ld,
                                                   int row0, int k0, int m16, int sel) {
    const _Float16* p = X + (size_t)(row0 + m16) * ld + k0;
    return combine16(ld8(p + 8 * sel), ld8(p + 16 + 8 * sel));
}

// B-fragment for B = W^T tile: column n = W row (col0+n), halves 0..7 = K k0+8*sel.., halves 8..15 = K k0+16+8*sel..
static __device__ __forceinline__ v16h load_b_frag(const _Float16* W, int ld,
                                                   int col0, int k0, int m16, int sel) {
    const _Float16* p = W + (size_t)(col0 + m16) * ld + k0;
    return combine16(ld8(p + 8 * sel), ld8(p + 16 + 8 * sel));
}

// ---------------------------------------------------------------------------
// 1) f32 -> f16 conversion
// ---------------------------------------------------------------------------
__global__ void cvt_f32_to_f16(const float* __restrict__ src,
                               _Float16* __restrict__ dst, int n) {
    int i = (blockIdx.x * blockDim.x + threadIdx.x) * 4;
    if (i + 3 < n) {
        float4 v = *(const float4*)(src + i);
        dst[i + 0] = (_Float16)v.x;
        dst[i + 1] = (_Float16)v.y;
        dst[i + 2] = (_Float16)v.z;
        dst[i + 3] = (_Float16)v.w;
    } else {
        for (int j = i; j < n; ++j) dst[j] = (_Float16)src[j];
    }
}

// ---------------------------------------------------------------------------
// 2) QKV projection GEMM: Y = X @ W^T + b   (X: 8192x1024 f16, W: 1024x1024 f16)
//    blockIdx.z: 0=Q (scaled by 1/sqrt(dk), -> (B,H,S,dk)), 1=K (-> (B,H,S,dk)),
//                2=V (-> transposed (B,H,dk,S))
//    WG = 256 thr = 8 waves; wave tile 32(M)x64(N); block tile 64x256.
// ---------------------------------------------------------------------------
__global__ void qkv_gemm(const _Float16* __restrict__ xh,
                         const _Float16* __restrict__ wqh,
                         const _Float16* __restrict__ wkh,
                         const _Float16* __restrict__ wvh,
                         const float* __restrict__ bq,
                         const float* __restrict__ bk,
                         const float* __restrict__ bv,
                         _Float16* __restrict__ Qo,
                         _Float16* __restrict__ Ko,
                         _Float16* __restrict__ Vt) {
    const int z    = blockIdx.z;
    const _Float16* W = (z == 0) ? wqh : (z == 1) ? wkh : wvh;
    const float*    bias = (z == 0) ? bq : (z == 1) ? bk : bv;

    const int tid  = threadIdx.x;
    const int wave = tid >> 5;
    const int lane = tid & 31;
    const int m16  = lane & 15;
    const int sel  = lane >> 4;
    const int wm   = wave & 1;       // 2 waves in M
    const int wn   = wave >> 1;      // 4 waves in N
    const int row0 = blockIdx.x * 64 + wm * 32;
    const int col0 = blockIdx.y * 256 + wn * 64;

    v8f acc[2][4];
#pragma unroll
    for (int i = 0; i < 2; ++i)
#pragma unroll
        for (int j = 0; j < 4; ++j) acc[i][j] = (v8f)(0.0f);

    for (int k0 = 0; k0 < D_MODEL; k0 += 32) {
        v16h a[2], b[4];
#pragma unroll
        for (int i = 0; i < 2; ++i)
            a[i] = load_a_frag(xh, D_MODEL, row0 + i * 16, k0, m16, sel);
#pragma unroll
        for (int j = 0; j < 4; ++j)
            b[j] = load_b_frag(W, D_MODEL, col0 + j * 16, k0, m16, sel);
#pragma unroll
        for (int i = 0; i < 2; ++i)
#pragma unroll
            for (int j = 0; j < 4; ++j)
                acc[i][j] = wmma_f16(a[i], b[j], acc[i][j]);
    }

    const float qscale = (z == 0) ? 0.125f : 1.0f;   // 1/sqrt(64) folded into Q
#pragma unroll
    for (int i = 0; i < 2; ++i) {
#pragma unroll
        for (int j = 0; j < 4; ++j) {
#pragma unroll
            for (int r = 0; r < 8; ++r) {
                int row = row0 + i * 16 + r + 8 * sel;   // token index
                int col = col0 + j * 16 + m16;           // feature index
                float v = (acc[i][j][r] + bias[col]) * qscale;
                _Float16 hv = (_Float16)v;
                int bb = row >> 11;          // /SEQ
                int ss = row & (SEQ - 1);
                int hh = col >> 6;           // /DK
                int cc = col & (DK - 1);
                if (z == 2) {
                    // V transposed: (B,H,dk,S)
                    Vt[((size_t)(bb * NHEAD + hh) * DK + cc) * SEQ + ss] = hv;
                } else {
                    _Float16* dst = (z == 0) ? Qo : Ko;
                    dst[((size_t)(bb * NHEAD + hh) * SEQ + ss) * DK + cc] = hv;
                }
            }
        }
    }
}

// ---------------------------------------------------------------------------
// 3) Flash attention: per wave 16 query rows, online softmax, keys in tiles of 32.
//    grid = (S/128, B*H); block = 256 (8 waves).
// ---------------------------------------------------------------------------
__global__ void attn_kernel(const _Float16* __restrict__ Qh,
                            const _Float16* __restrict__ Kh,
                            const _Float16* __restrict__ Vt,
                            _Float16* __restrict__ ctxh) {
    __shared__ _Float16 lds_p[8][16 * 32];   // per-wave private P tile (8 KB)

    const int tid  = threadIdx.x;
    const int wave = tid >> 5;
    const int lane = tid & 31;
    const int m16  = lane & 15;
    const int sel  = lane >> 4;

    const int bh = blockIdx.y;               // b*NHEAD + h
    const int b  = bh >> 4;
    const int h  = bh & 15;
    const int q0 = blockIdx.x * 128 + wave * 16;

    const _Float16* Qbase = Qh + (size_t)bh * SEQ * DK;
    const _Float16* Kbase = Kh + (size_t)bh * SEQ * DK;
    const _Float16* Vbase = Vt + (size_t)bh * DK * SEQ;

    // Q A-fragments (16 x 64 split into two 16x32 k-steps), loaded once.
    const _Float16* qrow = Qbase + (size_t)(q0 + m16) * DK;
    v16h aq0 = combine16(ld8(qrow + 8 * sel),      ld8(qrow + 16 + 8 * sel));
    v16h aq1 = combine16(ld8(qrow + 32 + 8 * sel), ld8(qrow + 48 + 8 * sel));

    v8f acc[4];
#pragma unroll
    for (int j = 0; j < 4; ++j) acc[j] = (v8f)(0.0f);
    float m_r[8], l_r[8];
#pragma unroll
    for (int r = 0; r < 8; ++r) { m_r[r] = -3.0e38f; l_r[r] = 0.0f; }

    _Float16* lp = &lds_p[wave][0];

    for (int kt = 0; kt < SEQ; kt += 32) {
        // ---- scores: 16 queries x 32 keys (already scaled; Q carries 1/8) ----
        v8f cs[2];
#pragma unroll
        for (int sub = 0; sub < 2; ++sub) {
            const _Float16* kr = Kbase + (size_t)(kt + sub * 16 + m16) * DK;
            v16h b0 = combine16(ld8(kr + 8 * sel),      ld8(kr + 16 + 8 * sel));
            v16h b1 = combine16(ld8(kr + 32 + 8 * sel), ld8(kr + 48 + 8 * sel));
            v8f c = (v8f)(0.0f);
            c = wmma_f16(aq0, b0, c);
            c = wmma_f16(aq1, b1, c);
            cs[sub] = c;
        }

        // ---- online softmax update (per-row stats live at (r, lane-half)) ----
#pragma unroll
        for (int r = 0; r < 8; ++r) {
            float t = fmaxf(cs[0][r], cs[1][r]);
            t = fmaxf(t, __shfl_xor(t, 1));
            t = fmaxf(t, __shfl_xor(t, 2));
            t = fmaxf(t, __shfl_xor(t, 4));
            t = fmaxf(t, __shfl_xor(t, 8));
            float mn = fmaxf(m_r[r], t);
            float sc = __expf(m_r[r] - mn);
            m_r[r] = mn;
            float p0 = __expf(cs[0][r] - mn);
            float p1 = __expf(cs[1][r] - mn);
            cs[0][r] = p0;
            cs[1][r] = p1;
            float rs = p0 + p1;
            rs += __shfl_xor(rs, 1);
            rs += __shfl_xor(rs, 2);
            rs += __shfl_xor(rs, 4);
            rs += __shfl_xor(rs, 8);
            l_r[r] = l_r[r] * sc + rs;
#pragma unroll
            for (int j = 0; j < 4; ++j) acc[j][r] *= sc;
        }

        // ---- P: C-layout -> A-layout via per-wave LDS tile ----
#pragma unroll
        for (int sub = 0; sub < 2; ++sub)
#pragma unroll
            for (int r = 0; r < 8; ++r)
                lp[(r + 8 * sel) * 32 + sub * 16 + m16] = (_Float16)cs[sub][r];

        asm volatile("s_wait_dscnt 0" ::: "memory");

        const _Float16* prow = lp + m16 * 32;
        v16h ap = combine16(*(const v8h*)(prow + 8 * sel),
                            *(const v8h*)(prow + 16 + 8 * sel));

        // ---- acc += P(16x32) @ V(32keys x 64dk), V stored transposed ----
#pragma unroll
        for (int j = 0; j < 4; ++j) {
            const _Float16* vr = Vbase + (size_t)(j * 16 + m16) * SEQ + kt;
            v16h bv = combine16(ld8(vr + 8 * sel), ld8(vr + 16 + 8 * sel));
            acc[j] = wmma_f16(ap, bv, acc[j]);
        }
    }

    // ---- normalize and store ctx (B,S,D) as f16 ----
#pragma unroll
    for (int r = 0; r < 8; ++r) {
        float inv = 1.0f / l_r[r];
        int srow = q0 + r + 8 * sel;
#pragma unroll
        for (int j = 0; j < 4; ++j) {
            float v = acc[j][r] * inv;
            ctxh[((size_t)b * SEQ + srow) * D_MODEL + h * DK + j * 16 + m16] = (_Float16)v;
        }
    }
}

// ---------------------------------------------------------------------------
// 4) Output projection + bias + residual: y = ctx @ Wo^T + bo + x  (f32 out)
// ---------------------------------------------------------------------------
__global__ void out_gemm(const _Float16* __restrict__ ctxh,
                         const _Float16* __restrict__ woh,
                         const float* __restrict__ bo,
                         const float* __restrict__ xres,
                         float* __restrict__ y) {
    const int tid  = threadIdx.x;
    const int wave = tid >> 5;
    const int lane = tid & 31;
    const int m16  = lane & 15;
    const int sel  = lane >> 4;
    const int wm   = wave & 1;
    const int wn   = wave >> 1;
    const int row0 = blockIdx.x * 64 + wm * 32;
    const int col0 = blockIdx.y * 256 + wn * 64;

    v8f acc[2][4];
#pragma unroll
    for (int i = 0; i < 2; ++i)
#pragma unroll
        for (int j = 0; j < 4; ++j) acc[i][j] = (v8f)(0.0f);

    for (int k0 = 0; k0 < D_MODEL; k0 += 32) {
        v16h a[2], b[4];
#pragma unroll
        for (int i = 0; i < 2; ++i)
            a[i] = load_a_frag(ctxh, D_MODEL, row0 + i * 16, k0, m16, sel);
#pragma unroll
        for (int j = 0; j < 4; ++j)
            b[j] = load_b_frag(woh, D_MODEL, col0 + j * 16, k0, m16, sel);
#pragma unroll
        for (int i = 0; i < 2; ++i)
#pragma unroll
            for (int j = 0; j < 4; ++j)
                acc[i][j] = wmma_f16(a[i], b[j], acc[i][j]);
    }

#pragma unroll
    for (int i = 0; i < 2; ++i)
#pragma unroll
        for (int j = 0; j < 4; ++j)
#pragma unroll
            for (int r = 0; r < 8; ++r) {
                int row = row0 + i * 16 + r + 8 * sel;
                int col = col0 + j * 16 + m16;
                size_t idx = (size_t)row * D_MODEL + col;
                y[idx] = acc[i][j][r] + bo[col] + xres[idx];
            }
}

// ---------------------------------------------------------------------------
// 5) LayerNorm in place over rows of 1024 (one block per row).
// ---------------------------------------------------------------------------
__global__ void ln_kernel(float* __restrict__ y,
                          const float* __restrict__ gamma,
                          const float* __restrict__ beta) {
    __shared__ float s_sum[256];
    __shared__ float s_sq[256];
    const int row = blockIdx.x;
    const int tid = threadIdx.x;
    float* rp = y + (size_t)row * D_MODEL;

    float4 v = *(const float4*)(rp + tid * 4);
    float sum = v.x + v.y + v.z + v.w;
    float sq  = v.x * v.x + v.y * v.y + v.z * v.z + v.w * v.w;
    s_sum[tid] = sum;
    s_sq[tid]  = sq;
    __syncthreads();
#pragma unroll
    for (int s = 128; s > 0; s >>= 1) {
        if (tid < s) {
            s_sum[tid] += s_sum[tid + s];
            s_sq[tid]  += s_sq[tid + s];
        }
        __syncthreads();
    }
    float mu  = s_sum[0] * (1.0f / D_MODEL);
    float var = s_sq[0] * (1.0f / D_MODEL) - mu * mu;
    float inv = rsqrtf(var + 1e-5f);

    float4 g = *(const float4*)(gamma + tid * 4);
    float4 bta = *(const float4*)(beta + tid * 4);
    float4 o;
    o.x = (v.x - mu) * inv * g.x + bta.x;
    o.y = (v.y - mu) * inv * g.y + bta.y;
    o.z = (v.z - mu) * inv * g.z + bta.z;
    o.w = (v.w - mu) * inv * g.w + bta.w;
    *(float4*)(rp + tid * 4) = o;
}

// ---------------------------------------------------------------------------
// Host launcher
// ---------------------------------------------------------------------------
extern "C" void kernel_launch(void* const* d_in, const int* in_sizes, int n_in,
                              void* d_out, int out_size, void* d_ws, size_t ws_size,
                              hipStream_t stream) {
    const float* x     = (const float*)d_in[0];
    const float* Wq    = (const float*)d_in[1];
    const float* bq    = (const float*)d_in[2];
    const float* Wk    = (const float*)d_in[3];
    const float* bk    = (const float*)d_in[4];
    const float* Wv    = (const float*)d_in[5];
    const float* bv    = (const float*)d_in[6];
    const float* Wo    = (const float*)d_in[7];
    const float* bo    = (const float*)d_in[8];
    const float* gamma = (const float*)d_in[9];
    const float* beta  = (const float*)d_in[10];
    float* y = (float*)d_out;

    // Workspace layout (~88 MB, all offsets 256B-aligned)
    char* ws = (char*)d_ws;
    _Float16* xh   = (_Float16*)(ws);                       // 16 MB
    _Float16* wqh  = (_Float16*)(ws + (size_t)16777216);    //  2 MB
    _Float16* wkh  = (_Float16*)(ws + (size_t)18874368);
    _Float16* wvh  = (_Float16*)(ws + (size_t)20971520);
    _Float16* woh  = (_Float16*)(ws + (size_t)23068672);
    _Float16* Qh   = (_Float16*)(ws + (size_t)25165824);    // 16 MB (B,H,S,dk)
    _Float16* Kh   = (_Float16*)(ws + (size_t)41943040);    // 16 MB (B,H,S,dk)
    _Float16* Vth  = (_Float16*)(ws + (size_t)58720256);    // 16 MB (B,H,dk,S)
    _Float16* ctxh = (_Float16*)(ws + (size_t)75497472);    // 16 MB (B,S,D)

    // 1) conversions
    cvt_f32_to_f16<<<8192, 256, 0, stream>>>(x,  xh,  NTOK * D_MODEL);
    cvt_f32_to_f16<<<1024, 256, 0, stream>>>(Wq, wqh, D_MODEL * D_MODEL);
    cvt_f32_to_f16<<<1024, 256, 0, stream>>>(Wk, wkh, D_MODEL * D_MODEL);
    cvt_f32_to_f16<<<1024, 256, 0, stream>>>(Wv, wvh, D_MODEL * D_MODEL);
    cvt_f32_to_f16<<<1024, 256, 0, stream>>>(Wo, woh, D_MODEL * D_MODEL);

    // 2) QKV projections (z: 0=Q,1=K,2=V)
    qkv_gemm<<<dim3(NTOK / 64, D_MODEL / 256, 3), 256, 0, stream>>>(
        xh, wqh, wkh, wvh, bq, bk, bv, Qh, Kh, Vth);

    // 3) flash attention
    attn_kernel<<<dim3(SEQ / 128, BATCH * NHEAD), 256, 0, stream>>>(Qh, Kh, Vth, ctxh);

    // 4) output projection + residual
    out_gemm<<<dim3(NTOK / 64, D_MODEL / 256), 256, 0, stream>>>(ctxh, woh, bo, x, y);

    // 5) layernorm in place
    ln_kernel<<<NTOK, 256, 0, stream>>>(y, gamma, beta);
}
